// HANLayer_60962765799924
// MI455X (gfx1250) — matrix operand order repaired and checked by reading, not compiled, not run on hardware
//
#include <hip/hip_runtime.h>
#include <hip/hip_bf16.h>
#include <stddef.h>

// ---------------- problem constants ----------------
#define NN   20000      // nodes
#define NE   640000     // edges per metapath
#define FIN  256        // input feature dim
#define NH   8          // heads
#define DD   16         // dim per head
#define NP   2          // metapaths
#define FH   128        // NH*DD
#define HID  128        // semantic hidden

typedef __attribute__((ext_vector_type(16))) __bf16 v16bf;
typedef __attribute__((ext_vector_type(8)))  __bf16 v8bf;
typedef __attribute__((ext_vector_type(8)))  float  v8f;

// order-preserving float<->uint for atomicMax-based segment max
__device__ __forceinline__ unsigned encf(float f) {
    unsigned b = __float_as_uint(f);
    return (b & 0x80000000u) ? ~b : (b | 0x80000000u);
}
__device__ __forceinline__ float decf(unsigned u) {
    unsigned b = (u & 0x80000000u) ? (u ^ 0x80000000u) : ~u;
    return __uint_as_float(b);
}
#define MU_SENTINEL 0x007FFFFFu   // encf(-inf)

// ---------------- init scratch ----------------
__global__ void k_init(float* __restrict__ agg, float* __restrict__ sb,
                       unsigned* __restrict__ mu, float* __restrict__ wsum) {
    int i = blockIdx.x * blockDim.x + threadIdx.x;
    if (i < NP * NN * FH) agg[i] = 0.0f;
    if (i < NP * NN * NH) { sb[i] = 0.0f; mu[i] = MU_SENTINEL; }
    if (i < NP) wsum[i] = 0.0f;
}

// ---------------- fp32 -> bf16 conversions ----------------
__global__ void k_conv_h(const float* __restrict__ h, __bf16* __restrict__ hb) {
    int i = blockIdx.x * blockDim.x + threadIdx.x;
    if (i < NN * FIN) hb[i] = (__bf16)h[i];
}
// W_gat [P, FIN, FH] -> Wt [P, FH, FIN] (transposed so B K-dim is contiguous)
__global__ void k_conv_w(const float* __restrict__ W, __bf16* __restrict__ wt) {
    int i = blockIdx.x * blockDim.x + threadIdx.x;
    if (i >= NP * FH * FIN) return;
    int p = i / (FH * FIN);
    int r = i - p * (FH * FIN);
    int n = r / FIN;
    int k = r - n * FIN;
    wt[i] = (__bf16)W[((size_t)p * FIN + k) * FH + n];
}

// ---------------- WMMA GEMM: z[p] = h @ W[p]  ([NN x FIN] x [FIN x FH]) ----------------
// one wave computes a 16-row x 128-col tile; 8 accumulators over the 8 N-tiles,
// K loop over FIN in steps of 32 with v_wmma_f32_16x16x32_bf16.
__global__ void __launch_bounds__(128) k_gemm(const __bf16* __restrict__ hb,
                                              const __bf16* __restrict__ wt,
                                              float* __restrict__ z) {
    const int MT = NN / 16;  // 1250 row tiles
    int w    = (blockIdx.x * 128 + threadIdx.x) >> 5;
    int lane = threadIdx.x & 31;
    if (w >= NP * MT) return;                 // wave-uniform
    int p    = w / MT;
    int m0   = (w - p * MT) * 16;
    int l15  = lane & 15;
    int half = lane >> 4;

    const __bf16* A     = hb + (size_t)(m0 + l15) * FIN;
    const __bf16* Bbase = wt + (size_t)p * FH * FIN + (size_t)l15 * FIN + half * 16;

    v8f acc[8] = {};
    for (int k0 = 0; k0 < FIN; k0 += 32) {
        // A layout: lanes 0-15 rows 0-15 / K 0..15; lanes 16-31 same rows, K 16..31
        union { v16bf v; v8bf h2[2]; } ua;
        ua.h2[0] = *(const v8bf*)(A + k0 + half * 16);
        ua.h2[1] = *(const v8bf*)(A + k0 + half * 16 + 8);
        v16bf a = ua.v;
#pragma unroll
        for (int t = 0; t < 8; t++) {
            // B layout: lanes 0-15 K=k0..k0+15, lanes 16-31 K=k0+16..k0+31, col = 16t + l15
            const __bf16* Bp = Bbase + (size_t)t * 16 * FIN + k0;
            union { v16bf v; v8bf h2[2]; } ub;
            ub.h2[0] = *(const v8bf*)(Bp);
            ub.h2[1] = *(const v8bf*)(Bp + 8);
            acc[t] = __builtin_amdgcn_wmma_f32_16x16x32_bf16(
                false, a, false, ub.v, (short)0, acc[t], false, false);
        }
    }
    // C/D layout: lanes 0-15 VGPR j -> row j; lanes 16-31 -> row j+8; col = lane%16
    float* zo = z + ((size_t)p * NN + m0) * FH;
#pragma unroll
    for (int t = 0; t < 8; t++)
#pragma unroll
        for (int j = 0; j < 8; j++)
            zo[(size_t)(j + 8 * half) * FH + 16 * t + l15] = acc[t][j];
}

// ---------------- per-node head scores el/er ----------------
__global__ void k_elr(const float* __restrict__ z, const float* __restrict__ al,
                      const float* __restrict__ ar, float* __restrict__ el,
                      float* __restrict__ er) {
    int i = blockIdx.x * blockDim.x + threadIdx.x;
    if (i >= NP * NN * NH) return;
    int p = i / (NN * NH);
    int r = i - p * (NN * NH);
    int n = r / NH;
    int h = r - n * NH;
    const float* zr  = z + ((size_t)p * NN + n) * FH + h * DD;
    const float* alp = al + (p * NH + h) * DD;
    const float* arp = ar + (p * NH + h) * DD;
    float sl = 0.0f, sr = 0.0f;
#pragma unroll
    for (int d = 0; d < DD; d++) {
        sl = fmaf(zr[d], alp[d], sl);
        sr = fmaf(zr[d], arp[d], sr);
    }
    el[i] = sl;
    er[i] = sr;
}

// ---------------- segment max over dst (encoded uint atomicMax) ----------------
__global__ void k_max(const int* __restrict__ src, const int* __restrict__ dst,
                      const float* __restrict__ el, const float* __restrict__ er,
                      unsigned* __restrict__ mu) {
    int i = blockIdx.x * blockDim.x + threadIdx.x;
    if (i >= NP * NE * NH) return;
    int p = i / (NE * NH);
    int r = i - p * (NE * NH);
    int e = r / NH;
    int h = r - e * NH;
    int s = src[(size_t)p * NE + e];
    int d = dst[(size_t)p * NE + e];
    float v = el[((size_t)p * NN + s) * NH + h] + er[((size_t)p * NN + d) * NH + h];
    v = v > 0.0f ? v : 0.2f * v;   // leaky_relu(0.2)
    atomicMax(&mu[((size_t)p * NN + d) * NH + h], encf(v));
}

// decode max; non-finite (empty segment) -> 0, as in reference
__global__ void k_mfin(const unsigned* __restrict__ mu, float* __restrict__ mf) {
    int i = blockIdx.x * blockDim.x + threadIdx.x;
    if (i >= NP * NN * NH) return;
    float f = decf(mu[i]);
    unsigned b = __float_as_uint(f);
    if ((b & 0x7F800000u) == 0x7F800000u) f = 0.0f;
    mf[i] = f;
}

// ---------------- segment sum of exp(e - m) ----------------
__global__ void k_sum(const int* __restrict__ src, const int* __restrict__ dst,
                      const float* __restrict__ el, const float* __restrict__ er,
                      const float* __restrict__ mf, float* __restrict__ sb) {
    int i = blockIdx.x * blockDim.x + threadIdx.x;
    if (i >= NP * NE * NH) return;
    int p = i / (NE * NH);
    int r = i - p * (NE * NH);
    int e = r / NH;
    int h = r - e * NH;
    int s = src[(size_t)p * NE + e];
    int d = dst[(size_t)p * NE + e];
    size_t di = ((size_t)p * NN + d) * NH + h;
    float v = el[((size_t)p * NN + s) * NH + h] + er[di];
    v = v > 0.0f ? v : 0.2f * v;
    atomicAdd(&sb[di], __expf(v - mf[di]));
}

// ---------------- weighted aggregation: one wave per edge ----------------
// lane owns 4 contiguous floats of the 128-wide row; head = lane>>2.
__global__ void __launch_bounds__(256) k_agg(const int* __restrict__ src,
                                             const int* __restrict__ dst,
                                             const float* __restrict__ el,
                                             const float* __restrict__ er,
                                             const float* __restrict__ mf,
                                             const float* __restrict__ sb,
                                             const float* __restrict__ z,
                                             float* __restrict__ agg) {
    int w    = (blockIdx.x * 256 + threadIdx.x) >> 5;
    int lane = threadIdx.x & 31;
    if (w >= NP * NE) return;
    int p = w / NE;
    int e = w - p * NE;
    int s = src[(size_t)p * NE + e];
    int d = dst[(size_t)p * NE + e];
    int h = lane >> 2;
    size_t di = ((size_t)p * NN + d) * NH + h;
    float v = el[((size_t)p * NN + s) * NH + h] + er[di];
    v = v > 0.0f ? v : 0.2f * v;
    float alpha = __expf(v - mf[di]) / (sb[di] + 1e-9f);
    const float4 zv = *(const float4*)(z + ((size_t)p * NN + s) * FH + lane * 4);
    float* o = agg + ((size_t)p * NN + d) * FH + lane * 4;
    atomicAdd(o + 0, alpha * zv.x);
    atomicAdd(o + 1, alpha * zv.y);
    atomicAdd(o + 2, alpha * zv.z);
    atomicAdd(o + 3, alpha * zv.w);
}

// ---------------- ELU in place ----------------
__global__ void k_elu(float* __restrict__ agg) {
    int i = blockIdx.x * blockDim.x + threadIdx.x;
    if (i >= NP * NN * FH) return;
    float x = agg[i];
    agg[i] = x > 0.0f ? x : (__expf(x) - 1.0f);
}

// ---------------- semantic attention logits: wsum[p] = sum_n tanh(z_n@W1+b1)@W2 ----------------
__global__ void __launch_bounds__(128) k_sem(const float* __restrict__ agg,
                                             const float* __restrict__ W1,
                                             const float* __restrict__ b1,
                                             const float* __restrict__ W2,
                                             float* __restrict__ wsum) {
    int pn = blockIdx.x;             // 0 .. NP*NN-1
    int j  = threadIdx.x;            // hidden unit
    const float* zr = agg + (size_t)pn * FH;
    float acc = b1[j];
    for (int k = 0; k < FH; k++) acc = fmaf(zr[k], W1[(size_t)k * HID + j], acc);
    float val = tanhf(acc) * W2[j];
    __shared__ float red[128];
    red[j] = val;
    __syncthreads();
    for (int s2 = 64; s2 > 0; s2 >>= 1) {
        if (j < s2) red[j] += red[j + s2];
        __syncthreads();
    }
    if (j == 0) atomicAdd(&wsum[pn / NN], red[0]);
}

// ---------------- softmax over metapaths + combine ----------------
__global__ void k_final(const float* __restrict__ agg, const float* __restrict__ wsum,
                        float* __restrict__ out) {
    int i = blockIdx.x * blockDim.x + threadIdx.x;
    if (i >= NN * FH) return;
    float w0 = wsum[0] * (1.0f / NN);
    float w1 = wsum[1] * (1.0f / NN);
    float mx = fmaxf(w0, w1);
    float e0 = __expf(w0 - mx), e1 = __expf(w1 - mx);
    float inv = 1.0f / (e0 + e1);
    out[i] = (e0 * agg[i] + e1 * agg[(size_t)NN * FH + i]) * inv;
}

// ---------------- launcher ----------------
extern "C" void kernel_launch(void* const* d_in, const int* in_sizes, int n_in,
                              void* d_out, int out_size, void* d_ws, size_t ws_size,
                              hipStream_t stream) {
    const float* h     = (const float*)d_in[0];
    const int*   src   = (const int*)d_in[1];
    const int*   dst   = (const int*)d_in[2];
    const float* W_gat = (const float*)d_in[3];
    const float* al    = (const float*)d_in[4];
    const float* ar    = (const float*)d_in[5];
    const float* W1    = (const float*)d_in[6];
    const float* b1    = (const float*)d_in[7];
    const float* W2    = (const float*)d_in[8];
    float* out = (float*)d_out;

    // workspace layout (all offsets 256B aligned)
    char* ws = (char*)d_ws;
    const size_t Z_BYTES   = (size_t)NP * NN * FH * 4;   // 20,480,000
    const size_t AGG_BYTES = Z_BYTES;
    const size_t HB_BYTES  = (size_t)NN * FIN * 2;       // 10,240,000
    const size_t WT_BYTES  = (size_t)NP * FH * FIN * 2;  // 131,072
    const size_t PH_BYTES  = (size_t)NP * NN * NH * 4;   // 1,280,000

    size_t off = 0;
    float*    z_buf  = (float*)(ws + off);    off += Z_BYTES;
    float*    agg    = (float*)(ws + off);    off += AGG_BYTES;
    __bf16*   hb     = (__bf16*)(ws + off);   off += HB_BYTES;
    __bf16*   wt     = (__bf16*)(ws + off);   off += WT_BYTES;
    float*    el     = (float*)(ws + off);    off += PH_BYTES;
    float*    er     = (float*)(ws + off);    off += PH_BYTES;
    unsigned* mu     = (unsigned*)(ws + off); off += PH_BYTES;
    float*    mf     = (float*)(ws + off);    off += PH_BYTES;
    float*    sb     = (float*)(ws + off);    off += PH_BYTES;
    float*    wsum   = (float*)(ws + off);    off += 256;

    const int T = 256;
    // init scratch (agg zero, segment-sum zero, segment-max sentinel, logits zero)
    k_init<<<(NP * NN * FH + T - 1) / T, T, 0, stream>>>(agg, sb, mu, wsum);
    // bf16 conversions
    k_conv_h<<<(NN * FIN + T - 1) / T, T, 0, stream>>>(h, hb);
    k_conv_w<<<(NP * FH * FIN + T - 1) / T, T, 0, stream>>>(W_gat, wt);
    // WMMA GEMM: 2500 waves, 4 waves/block
    k_gemm<<<(NP * (NN / 16) + 3) / 4, 128, 0, stream>>>(hb, wt, z_buf);
    // per-node head scores
    k_elr<<<(NP * NN * NH + T - 1) / T, T, 0, stream>>>(z_buf, al, ar, el, er);
    // edge softmax (max, finalize, sum)
    k_max<<<(NP * NE * NH + T - 1) / T, T, 0, stream>>>(src, dst, el, er, mu);
    k_mfin<<<(NP * NN * NH + T - 1) / T, T, 0, stream>>>(mu, mf);
    k_sum<<<(NP * NE * NH + T - 1) / T, T, 0, stream>>>(src, dst, el, er, mf, sb);
    // weighted aggregation: one wave per edge
    k_agg<<<(NP * NE + 7) / 8, 256, 0, stream>>>(src, dst, el, er, mf, sb, z_buf, agg);
    // ELU
    k_elu<<<(NP * NN * FH + T - 1) / T, T, 0, stream>>>(agg);
    // semantic attention logits
    k_sem<<<NP * NN, 128, 0, stream>>>(agg, W1, b1, W2, wsum);
    // softmax over metapaths + combine into output
    k_final<<<(NN * FH + T - 1) / T, T, 0, stream>>>(agg, wsum, out);
}
